// ModulatedDCN2dBLKv1_72524817760946
// MI455X (gfx1250) — compile-verified, matrix operand order checked by reading, not run on hardware
//
#include <hip/hip_runtime.h>
#include <hip/hip_bf16.h>

#define C_IN 256
#define C_OUT 256
#define KK 9
#define STYLE_DIM 512
#define Hh 64
#define Ww 64
#define HW 4096          // 64*64
#define KDIM (C_IN * KK) // 2304
#define NB 4
#define KSLICES (KDIM / 32) // 72

typedef __attribute__((ext_vector_type(16))) _Float16 v16h;
typedef __attribute__((ext_vector_type(8)))  _Float16 v8h;
typedef __attribute__((ext_vector_type(8)))  float    v8f;

union Frag16 { v16h v; v8h h[2]; };

static __device__ __forceinline__ int clampi(int v, int lo, int hi) {
    return v < lo ? lo : (v > hi ? hi : v);
}

// ---------------------------------------------------------------------------
// Kernel 1: s[b,c] = style[b] . (mod_w[c] * MOD_SCALE) + mod_b[c]
// ---------------------------------------------------------------------------
__global__ void __launch_bounds__(C_IN)
style_mod_kernel(const float* __restrict__ style, const float* __restrict__ mod_w,
                 const float* __restrict__ mod_b, float* __restrict__ s)
{
    const float MOD_SCALE = 0.04419417382415922f; // 1/sqrt(512)
    int b = blockIdx.x, c = threadIdx.x;
    const float* st = style + (size_t)b * STYLE_DIM;
    const float* mw = mod_w + (size_t)c * STYLE_DIM;
    float acc = mod_b[c];
    for (int d = 0; d < STYLE_DIM; ++d) acc += st[d] * mw[d] * MOD_SCALE;
    s[b * C_IN + c] = acc;
}

// ---------------------------------------------------------------------------
// Kernel 2: modulated + demodulated weights, f16, N-major with kk-major K:
//   Bm[b][o][kk*256 + c]
// ---------------------------------------------------------------------------
__global__ void __launch_bounds__(C_IN)
weight_mod_kernel(const float* __restrict__ weight, const float* __restrict__ s,
                  _Float16* __restrict__ Bm)
{
    const float SCALE = 0.02083333333333333f; // 1/sqrt(256*9) = 1/48
    int o = blockIdx.x, b = blockIdx.y, c = threadIdx.x;
    float sc = s[b * C_IN + c];
    const float* wp = weight + ((size_t)o * C_IN + c) * KK;
    float w9[KK];
    float ss = 0.f;
    #pragma unroll
    for (int k = 0; k < KK; ++k) { float v = SCALE * wp[k] * sc; w9[k] = v; ss += v * v; }
    __shared__ float red[C_IN];
    red[c] = ss;
    __syncthreads();
    for (int st = C_IN / 2; st > 0; st >>= 1) {
        if (c < st) red[c] += red[c + st];
        __syncthreads();
    }
    float demod = rsqrtf(red[0] + 1e-8f);
    _Float16* bp = Bm + ((size_t)b * C_OUT + o) * KDIM + c; // + kk*256 below
    #pragma unroll
    for (int k = 0; k < KK; ++k) bp[k * C_IN] = (_Float16)(w9[k] * demod);
}

// ---------------------------------------------------------------------------
// Kernel 3: deformable bilinear im2col -> f16 A[b][p][kk*256 + c]
// One thread per (b, pixel); tap geometry once per tap; v8h (16B) stores.
// ---------------------------------------------------------------------------
__global__ void __launch_bounds__(256)
im2col_kernel(const float* __restrict__ input, const float* __restrict__ offset,
              const float* __restrict__ mask, _Float16* __restrict__ A)
{
    int b = blockIdx.y;
    int p = blockIdx.x * blockDim.x + threadIdx.x; // 0..4095
    int h = p >> 6, w = p & 63;
    const float* inb = input + (size_t)b * C_IN * HW;
    _Float16* Arow = A + ((size_t)b * HW + p) * KDIM;
    for (int kk = 0; kk < KK; ++kk) {
        float offy = offset[(((size_t)b * KK + kk) * 2 + 0) * HW + p];
        float offx = offset[(((size_t)b * KK + kk) * 2 + 1) * HW + p];
        float mv   = mask[((size_t)b * KK + kk) * HW + p];
        float py = (float)(h + (kk / 3) - 1) + offy;
        float px = (float)(w + (kk % 3) - 1) + offx;
        float y0f = floorf(py), x0f = floorf(px);
        float dy = py - y0f, dx = px - x0f;
        int y0 = (int)y0f, x0 = (int)x0f;
        int y1 = y0 + 1,  x1 = x0 + 1;
        int y0c = clampi(y0, 0, Hh - 1), y1c = clampi(y1, 0, Hh - 1);
        int x0c = clampi(x0, 0, Ww - 1), x1c = clampi(x1, 0, Ww - 1);
        float vy0 = (y0 >= 0 && y0 < Hh) ? 1.f : 0.f;
        float vy1 = (y1 >= 0 && y1 < Hh) ? 1.f : 0.f;
        float vx0 = (x0 >= 0 && x0 < Ww) ? 1.f : 0.f;
        float vx1 = (x1 >= 0 && x1 < Ww) ? 1.f : 0.f;
        float w00 = (1.f - dy) * (1.f - dx) * vy0 * vx0 * mv;
        float w01 = (1.f - dy) * dx        * vy0 * vx1 * mv;
        float w10 = dy * (1.f - dx)        * vy1 * vx0 * mv;
        float w11 = dy * dx                * vy1 * vx1 * mv;
        int i00 = y0c * Ww + x0c, i01 = y0c * Ww + x1c;
        int i10 = y1c * Ww + x0c, i11 = y1c * Ww + x1c;
        _Float16* dst = Arow + kk * C_IN;
        for (int c0 = 0; c0 < C_IN; c0 += 8) {
            v8h buf;
            #pragma unroll
            for (int u = 0; u < 8; ++u) {
                const float* cp = inb + (size_t)(c0 + u) * HW;
                float v = w00 * cp[i00] + w01 * cp[i01] + w10 * cp[i10] + w11 * cp[i11];
                buf[u] = (_Float16)v;
            }
            *(v8h*)(dst + c0) = buf;
        }
    }
}

// ---------------------------------------------------------------------------
// GEMM helpers: load one 32-wide K-slice of fragments / issue its 8 WMMAs.
// ---------------------------------------------------------------------------
static __device__ __forceinline__ void
load_slice(const _Float16* const (&Ap)[2], const _Float16* const (&Bp)[4],
           int kg, Frag16 (&fa)[2], Frag16 (&fb)[4])
{
    #pragma unroll
    for (int i = 0; i < 2; ++i) {
        fa[i].h[0] = *(const v8h*)(Ap[i] + kg);
        fa[i].h[1] = *(const v8h*)(Ap[i] + kg + 16);
    }
    #pragma unroll
    for (int j = 0; j < 4; ++j) {
        fb[j].h[0] = *(const v8h*)(Bp[j] + kg);
        fb[j].h[1] = *(const v8h*)(Bp[j] + kg + 16);
    }
}

static __device__ __forceinline__ void
wmma_slice(const Frag16 (&fa)[2], const Frag16 (&fb)[4], v8f (&acc)[2][4])
{
    #pragma unroll
    for (int i = 0; i < 2; ++i)
        #pragma unroll
        for (int j = 0; j < 4; ++j)
            acc[i][j] = __builtin_amdgcn_wmma_f32_16x16x32_f16(
                false, fa[i].v, false, fb[j].v, (short)0, acc[i][j], false, false);
}

// ---------------------------------------------------------------------------
// Kernel 4: per-batch GEMM via v_wmma_f32_16x16x32_f16 + fused bias/leaky^2.
// Block = 8 waves as 2(M) x 4(N); wave tile = 32(M) x 64(N) -> 8 accumulators.
// Block tile = 64(M) x 256(N) (full N). Explicit depth-2 ping-pong pipeline;
// further compiler unrolling is disabled to keep VGPR liveness bounded
// (prevents accumulator shuffle copies seen at 4-slices-in-flight).
// ---------------------------------------------------------------------------
__global__ void __launch_bounds__(256)
dcn_wmma_gemm_kernel(const _Float16* __restrict__ A, const _Float16* __restrict__ Bm,
                     const float* __restrict__ bias1, const float* __restrict__ bias2,
                     float* __restrict__ mid)
{
    const float SQRT2 = 1.4142135623730951f;
    int b      = blockIdx.z;
    int lane   = threadIdx.x & 31;
    int wave   = threadIdx.x >> 5;
    int waveN  = wave & 3;          // 0..3 -> N offset 0,64,128,192
    int waveM  = wave >> 2;         // 0..1 -> M offset 0,32
    int lanelo = lane & 15;
    int kb     = (lane >> 4) << 3;  // K sub-offset (0 or 8)
    int mTile  = blockIdx.x * 64 + waveM * 32;
    int nTile  = waveN * 64;

    const _Float16* Ap[2];
    const _Float16* Bp[4];
    #pragma unroll
    for (int i = 0; i < 2; ++i)
        Ap[i] = A + ((size_t)b * HW + mTile + i * 16 + lanelo) * KDIM + kb;
    #pragma unroll
    for (int j = 0; j < 4; ++j)
        Bp[j] = Bm + ((size_t)b * C_OUT + nTile + j * 16 + lanelo) * KDIM + kb;

    // one-shot warm-up prefetch of the upcoming A/B lines
    __builtin_prefetch(Ap[0], 0, 3);
    __builtin_prefetch(Bp[0], 0, 3);

    v8f acc[2][4];
    #pragma unroll
    for (int i = 0; i < 2; ++i)
        #pragma unroll
        for (int j = 0; j < 4; ++j) acc[i][j] = (v8f){};

    Frag16 a0[2], b0[4], a1[2], b1[4];      // ping-pong buffers
    load_slice(Ap, Bp, 0, a0, b0);          // slice 0

    // main pipeline: slices 0..69 computed, 1..70 loaded (35 iterations)
    #pragma clang loop unroll(disable)
    for (int t = 0; t < KSLICES - 2; t += 2) {
        load_slice(Ap, Bp, (t + 1) * 32, a1, b1);
        wmma_slice(a0, b0, acc);
        load_slice(Ap, Bp, (t + 2) * 32, a0, b0);
        wmma_slice(a1, b1, acc);
    }
    // tail: slice 70 is in buf0; load 71, drain both
    load_slice(Ap, Bp, (KSLICES - 1) * 32, a1, b1);
    wmma_slice(a0, b0, acc);
    wmma_slice(a1, b1, acc);

    // C/D layout: VGPR r, lane<16 -> M=r ; lane>=16 -> M=r+8 ; N = lane&15.
    int mrow = mTile + ((lane >> 4) << 3);
    #pragma unroll
    for (int j = 0; j < 4; ++j) {
        int o = nTile + j * 16 + lanelo;
        float b1v = bias1[o], b2v = bias2[o];
        float* op = mid + ((size_t)b * C_OUT + o) * HW + mrow;
        #pragma unroll
        for (int i = 0; i < 2; ++i) {
            #pragma unroll
            for (int r = 0; r < 8; ++r) {
                float v = acc[i][j][r] + b1v;
                v = (v > 0.f ? v : 0.2f * v) * SQRT2;
                v = v + b2v;
                v = (v > 0.f ? v : 0.2f * v) * SQRT2;
                op[i * 16 + r] = v;
            }
        }
    }
}

// ---------------------------------------------------------------------------
// Kernel 5: upfirdn2d up=2, FIR [1,3,3,1] (x4 gain) -> 2x2 taps per out pixel.
// ---------------------------------------------------------------------------
__global__ void __launch_bounds__(256)
upfirdn_kernel(const float* __restrict__ mid, float* __restrict__ out)
{
    size_t t = (size_t)blockIdx.x * 256 + threadIdx.x; // [b][o][y][x], 4*256*128*128
    int x = (int)(t & 127);
    int y = (int)((t >> 7) & 127);
    size_t bo = t >> 14;
    const float* mp = mid + bo * HW;
    int iy0, iy1, ix0, ix1;
    float wy0, wy1, wx0, wx1;
    if ((y & 1) == 0) { iy0 = (y >> 1) - 1; wy0 = 1.f; iy1 = y >> 1;       wy1 = 3.f; }
    else              { iy0 = (y - 1) >> 1; wy0 = 3.f; iy1 = (y + 1) >> 1; wy1 = 1.f; }
    if ((x & 1) == 0) { ix0 = (x >> 1) - 1; wx0 = 1.f; ix1 = x >> 1;       wx1 = 3.f; }
    else              { ix0 = (x - 1) >> 1; wx0 = 3.f; ix1 = (x + 1) >> 1; wx1 = 1.f; }
    float acc = 0.f;
    if (iy0 >= 0 && iy0 < Hh) {
        if (ix0 >= 0 && ix0 < Ww) acc += wy0 * wx0 * mp[iy0 * Ww + ix0];
        if (ix1 >= 0 && ix1 < Ww) acc += wy0 * wx1 * mp[iy0 * Ww + ix1];
    }
    if (iy1 >= 0 && iy1 < Hh) {
        if (ix0 >= 0 && ix0 < Ww) acc += wy1 * wx0 * mp[iy1 * Ww + ix0];
        if (ix1 >= 0 && ix1 < Ww) acc += wy1 * wx1 * mp[iy1 * Ww + ix1];
    }
    out[t] = acc * 0.0625f; // /16
}

// ---------------------------------------------------------------------------
extern "C" void kernel_launch(void* const* d_in, const int* in_sizes, int n_in,
                              void* d_out, int out_size, void* d_ws, size_t ws_size,
                              hipStream_t stream)
{
    (void)in_sizes; (void)n_in; (void)out_size; (void)ws_size;
    const float* input  = (const float*)d_in[0];
    const float* style  = (const float*)d_in[1];
    const float* offset = (const float*)d_in[2];
    const float* mask   = (const float*)d_in[3];
    const float* weight = (const float*)d_in[4];
    const float* mod_w  = (const float*)d_in[5];
    const float* mod_b  = (const float*)d_in[6];
    const float* bias1  = (const float*)d_in[7];
    const float* bias2  = (const float*)d_in[8];
    float* out = (float*)d_out;

    // Workspace layout (all 16B-aligned):
    //   s   : 4*256 f32                      @ 0          (4 KB, pad to 8 KB)
    //   Bm  : 4*256*2304 f16  (~4.5 MB)      @ 8192
    //   A   : 4*4096*2304 f16 (~72 MB)       @ 8192 + 4718592
    //   mid : 4*256*4096 f32  (16 MB)        @ ... + 75497472
    char* ws = (char*)d_ws;
    const size_t OFF_BM  = 8192;
    const size_t OFF_A   = OFF_BM + (size_t)NB * C_OUT * KDIM * sizeof(_Float16);
    const size_t OFF_MID = OFF_A  + (size_t)NB * HW * KDIM * sizeof(_Float16);
    float*    s   = (float*)ws;
    _Float16* Bm  = (_Float16*)(ws + OFF_BM);
    _Float16* A   = (_Float16*)(ws + OFF_A);
    float*    mid = (float*)(ws + OFF_MID);

    style_mod_kernel<<<dim3(NB), dim3(C_IN), 0, stream>>>(style, mod_w, mod_b, s);
    weight_mod_kernel<<<dim3(C_OUT, NB), dim3(C_IN), 0, stream>>>(weight, s, Bm);
    im2col_kernel<<<dim3(HW / 256, NB), dim3(256), 0, stream>>>(input, offset, mask, A);
    dcn_wmma_gemm_kernel<<<dim3(HW / 64, 1, NB), dim3(256), 0, stream>>>(A, Bm, bias1, bias2, mid);
    upfirdn_kernel<<<dim3((NB * C_OUT * 128 * 128) / 256), dim3(256), 0, stream>>>(mid, out);
}
